// GRU_47098611368316
// MI455X (gfx1250) — matrix operand order repaired and checked by reading, not compiled, not run on hardware
//
#include <hip/hip_runtime.h>
#include <hip/hip_bf16.h>
#include <cstdint>
#include <cstddef>

// Problem constants (match reference)
#define BQ   32
#define SQ   512
#define EQ   1024
#define HQ   2048
#define BSQ  (BQ * SQ)   // 16384 rows

typedef __attribute__((ext_vector_type(16))) __bf16        bf16x16;
typedef __attribute__((ext_vector_type(8)))  float         floatx8;
typedef __attribute__((ext_vector_type(4)))  unsigned int  uint4v;

union FragBF { bf16x16 f; uint4v u[2]; };

__device__ __forceinline__ floatx8 wmma_bf16(bf16x16 a, bf16x16 b, floatx8 c) {
  // D = A(16x32 bf16) * B(32x16 bf16) + C(16x16 f32)
  return __builtin_amdgcn_wmma_f32_16x16x32_bf16(false, a, false, b, (short)0, c,
                                                 false, false);
}

__device__ __forceinline__ float sigmoidf_(float x) {
  return 1.0f / (1.0f + __expf(-x));
}

// ---------------------------------------------------------------- utilities
__global__ void cvt_f32_bf16(const float* __restrict__ src,
                             __bf16* __restrict__ dst, long n) {
  long stride = (long)gridDim.x * blockDim.x;
  for (long i = (long)blockIdx.x * blockDim.x + threadIdx.x; i < n; i += stride)
    dst[i] = (__bf16)src[i];
}

// x[b*S+s, :] = bf16(emb[tokens[b,s], :])   (E == 1024)
__global__ void embed_gather(const int* __restrict__ tok,
                             const float* __restrict__ emb,
                             __bf16* __restrict__ x, long n) {
  long stride = (long)gridDim.x * blockDim.x;
  for (long i = (long)blockIdx.x * blockDim.x + threadIdx.x; i < n; i += stride) {
    long row = i >> 10;          // / EQ
    long col = i & (EQ - 1);
    x[i] = (__bf16)emb[(long)tok[row] * EQ + col];
  }
}

// ------------------------------------------------ batched gate GEMM (WMMA)
// C[M,N] (f32) = A[M,K](bf16, row-major) * W[N,K](bf16, row-major)^T + bias[N]
// block: 256 thr = 8 waves; block tile 64M x 256N; wave tile 32M x 64N.
// Double-buffered K loop: fragments for k+32 are loaded while WMMAs for k run.
// M % 64 == 0, N % 256 == 0, K % 32 == 0 (true for all three layers).
__global__ __launch_bounds__(256)
void wmma_gemm_bias(const __bf16* __restrict__ A, const __bf16* __restrict__ W,
                    const float* __restrict__ bias, float* __restrict__ C,
                    int M, int N, int K) {
  const int lane  = threadIdx.x & 31;
  const int wid   = threadIdx.x >> 5;
  const int mt    = wid & 1;               // which 32-row half of the 64-row tile
  const int ng    = wid >> 1;              // 0..3 -> 64-col group
  const int mBase = blockIdx.x * 64 + mt * 32;
  const int nBase = blockIdx.y * 256 + ng * 64;
  const int l15   = lane & 15;
  const int hi    = lane >> 4;             // 0 | 1
  const int koffA = hi * 8;                // A: lanes16-31 hold K 8..15 / 24..31
  const int koffB = hi * 8;                // B: lanes16-31 hold K 8..15 / 24..31

  const __bf16* aRow0 = A + (size_t)(mBase + l15) * K + koffA;       // rows 0..15
  const __bf16* aRow1 = aRow0 + (size_t)16 * K;                      // rows 16..31
  const __bf16* wRow[4];
#pragma unroll
  for (int f = 0; f < 4; ++f)
    wRow[f] = W + (size_t)(nBase + f * 16 + l15) * K + koffB;

  floatx8 acc[8] = {};

  auto loadStep = [&](int kb, FragBF& a0, FragBF& a1, FragBF* b) {
    a0.u[0] = *(const uint4v*)(aRow0 + kb);
    a0.u[1] = *(const uint4v*)(aRow0 + kb + 16);
    a1.u[0] = *(const uint4v*)(aRow1 + kb);
    a1.u[1] = *(const uint4v*)(aRow1 + kb + 16);
#pragma unroll
    for (int f = 0; f < 4; ++f) {
      b[f].u[0] = *(const uint4v*)(wRow[f] + kb);
      b[f].u[1] = *(const uint4v*)(wRow[f] + kb + 16);
    }
  };

  FragBF a0c, a1c, bc[4];
  loadStep(0, a0c, a1c, bc);

  for (int kb = 0; kb < K - 32; kb += 32) {
    FragBF a0n, a1n, bn[4];
    loadStep(kb + 32, a0n, a1n, bn);       // in flight while WMMAs execute
#pragma unroll
    for (int f = 0; f < 4; ++f) {
      acc[f]     = wmma_bf16(a0c.f, bc[f].f, acc[f]);
      acc[4 + f] = wmma_bf16(a1c.f, bc[f].f, acc[4 + f]);
    }
    a0c = a0n; a1c = a1n;
#pragma unroll
    for (int f = 0; f < 4; ++f) bc[f] = bn[f];
  }
#pragma unroll
  for (int f = 0; f < 4; ++f) {
    acc[f]     = wmma_bf16(a0c.f, bc[f].f, acc[f]);
    acc[4 + f] = wmma_bf16(a1c.f, bc[f].f, acc[4 + f]);
  }

#pragma unroll
  for (int f = 0; f < 4; ++f) {
    const int n = nBase + f * 16 + l15;
    const float bns = bias[n];
#pragma unroll
    for (int mh = 0; mh < 2; ++mh) {
#pragma unroll
      for (int e = 0; e < 8; ++e) {
        const int r = mBase + mh * 16 + e + hi * 8;   // C/D: VGPR e -> rows e / e+8
        C[(size_t)r * N + n] = acc[mh * 4 + f][e] + bns;
      }
    }
  }
}

// ------------------------------------------------------ grid-wide barrier
__device__ __forceinline__ void grid_barrier(unsigned int* bar, unsigned int nb) {
  __syncthreads();
  if (threadIdx.x == 0) {
    __threadfence();
    unsigned int gen = __hip_atomic_load(bar + 1, __ATOMIC_RELAXED,
                                         __HIP_MEMORY_SCOPE_AGENT);
    unsigned int t = __hip_atomic_fetch_add(bar, 1u, __ATOMIC_ACQ_REL,
                                            __HIP_MEMORY_SCOPE_AGENT);
    if (t == nb - 1u) {
      __hip_atomic_store(bar, 0u, __ATOMIC_RELAXED, __HIP_MEMORY_SCOPE_AGENT);
      __hip_atomic_store(bar + 1, gen + 1u, __ATOMIC_RELEASE,
                         __HIP_MEMORY_SCOPE_AGENT);
    } else {
      while (__hip_atomic_load(bar + 1, __ATOMIC_ACQUIRE,
                               __HIP_MEMORY_SCOPE_AGENT) == gen)
        __builtin_amdgcn_s_sleep(2);
    }
    __threadfence();
  }
  __syncthreads();
}

// ------------------------------------------------- persistent GRU scan
// One launch per layer. grid.x = H/32 blocks, 256 threads (8 waves).
// Block owns hidden units [j0, j0+32) across all 3 gates.
// LDS: h (32 x H bf16) + hg tile (3 x 32 x 32 f32).
__global__ __launch_bounds__(256)
void gru_scan(const __bf16* __restrict__ Whh,   // [3H, H] bf16
              const float*  __restrict__ bhh,   // [3H]
              const float*  __restrict__ xg,    // [B*S, 3H] f32 (b_ih included)
              float*        __restrict__ h,     // [32, H] f32   (init 0)
              __bf16*       __restrict__ hb,    // [32, H] bf16  (init 0)
              __bf16*       __restrict__ outb,  // [B*S, H] bf16 or null
              float*        __restrict__ outf,  // [B*S, H] f32  or null
              int H, int S, unsigned int* bar, unsigned int nb) {
  extern __shared__ char smem[];
  __bf16* hL = (__bf16*)smem;                          // [32][H]
  float*  gL = (float*)(smem + (size_t)BQ * H * 2);    // [3][32][32]

  const int lane  = threadIdx.x & 31;
  const int wid   = threadIdx.x >> 5;
  const int l15   = lane & 15;
  const int hi    = lane >> 4;
  const int koffA = hi * 8;
  const int koffB = hi * 8;
  const int j0    = blockIdx.x * 32;
  const int G3    = 3 * H;
  const int gate  = wid % 3;                 // valid for wid < 6
  const int jt    = wid / 3;                 // 0 | 1
  const long hChunks = (long)BQ * H / 8;     // uint4v chunks of h

  const __bf16* wp =
      Whh + (size_t)(gate * H + j0 + jt * 16 + l15) * H + koffB;
  const __bf16* h0 = hL + (size_t)l15 * H + koffA;     // rows 0..15
  const __bf16* h1 = h0 + (size_t)16 * H;              // rows 16..31

  for (int t = 0; t < S; ++t) {
    // stage full h (32 x H bf16) into LDS
    for (long i = threadIdx.x; i < hChunks; i += blockDim.x)
      ((uint4v*)hL)[i] = ((const uint4v*)hb)[i];
    __syncthreads();

    if (wid < 6) {
      // hg[0:32, n0:n0+16] for n0 = gate*H + j0 + jt*16
      floatx8 acc0 = {}, acc1 = {};
      FragBF bcur;
      bcur.u[0] = *(const uint4v*)(wp);
      bcur.u[1] = *(const uint4v*)(wp + 16);
      for (int kb = 0; kb < H - 32; kb += 32) {
        FragBF bnxt;                                   // W stream, double-buffered
        bnxt.u[0] = *(const uint4v*)(wp + kb + 32);
        bnxt.u[1] = *(const uint4v*)(wp + kb + 32 + 16);
        __builtin_prefetch(wp + kb + 1024, 0, 1);      // warm WGP$ (L2-resident W)
        FragBF a0, a1;
        a0.u[0] = *(const uint4v*)(h0 + kb);
        a0.u[1] = *(const uint4v*)(h0 + kb + 16);
        a1.u[0] = *(const uint4v*)(h1 + kb);
        a1.u[1] = *(const uint4v*)(h1 + kb + 16);
        acc0 = wmma_bf16(a0.f, bcur.f, acc0);
        acc1 = wmma_bf16(a1.f, bcur.f, acc1);
        bcur = bnxt;
      }
      {
        const int kb = H - 32;
        FragBF a0, a1;
        a0.u[0] = *(const uint4v*)(h0 + kb);
        a0.u[1] = *(const uint4v*)(h0 + kb + 16);
        a1.u[0] = *(const uint4v*)(h1 + kb);
        a1.u[1] = *(const uint4v*)(h1 + kb + 16);
        acc0 = wmma_bf16(a0.f, bcur.f, acc0);
        acc1 = wmma_bf16(a1.f, bcur.f, acc1);
      }
      const int cc = jt * 16 + l15;
#pragma unroll
      for (int e = 0; e < 8; ++e) {
        gL[(gate * 32 + (e + hi * 8)) * 32 + cc]      = acc0[e];
        gL[(gate * 32 + (16 + e + hi * 8)) * 32 + cc] = acc1[e];
      }
    }
    __syncthreads();

    // gate math + state update for this block's 32x32 (batch x unit) patch
    for (int i = threadIdx.x; i < BQ * 32; i += blockDim.x) {
      const int b  = i >> 5;
      const int jj = i & 31;
      const int j  = j0 + jj;
      const float* xrow = xg + ((size_t)b * S + t) * G3;
      const float xr = xrow[j];
      const float xz = xrow[H + j];
      const float xn = xrow[2 * H + j];
      const float hr = gL[(0 * 32 + b) * 32 + jj] + bhh[j];
      const float hz = gL[(1 * 32 + b) * 32 + jj] + bhh[H + j];
      const float hn = gL[(2 * 32 + b) * 32 + jj] + bhh[2 * H + j];
      const float r  = sigmoidf_(xr + hr);
      const float z  = sigmoidf_(xz + hz);
      const float nn = tanhf(xn + r * hn);
      const size_t hidx = (size_t)b * H + j;
      const float hnew = (1.0f - z) * nn + z * h[hidx];
      h[hidx]  = hnew;
      hb[hidx] = (__bf16)hnew;
      const size_t oidx = ((size_t)b * S + t) * H + j;
      if (outb) outb[oidx] = (__bf16)hnew;
      if (outf) outf[oidx] = hnew;
    }
    __syncthreads();
    grid_barrier(bar, nb);   // publish h before any block starts step t+1
  }
}

// ---------------------------------------------------------------- launcher
extern "C" void kernel_launch(void* const* d_in, const int* in_sizes, int n_in,
                              void* d_out, int out_size, void* d_ws, size_t ws_size,
                              hipStream_t stream) {
  (void)in_sizes; (void)n_in; (void)out_size; (void)ws_size;

  const int*   tokens = (const int*)d_in[0];
  const float* emb    = (const float*)d_in[1];
  const float* Wih[3] = {(const float*)d_in[2], (const float*)d_in[6],  (const float*)d_in[10]};
  const float* Whh[3] = {(const float*)d_in[3], (const float*)d_in[7],  (const float*)d_in[11]};
  const float* bih[3] = {(const float*)d_in[4], (const float*)d_in[8],  (const float*)d_in[12]};
  const float* bhh[3] = {(const float*)d_in[5], (const float*)d_in[9],  (const float*)d_in[13]};
  float* out = (float*)d_out;

  char* ws = (char*)d_ws;
  size_t off = 0;
  auto alloc = [&](size_t bytes) -> void* {
    off = (off + 255) & ~(size_t)255;
    void* p = ws + off;
    off += bytes;
    return p;
  };

  const int din [3] = {EQ, HQ, HQ};
  const int dout[3] = {HQ, HQ, EQ};

  unsigned int* bar = (unsigned int*)alloc(256);
  __bf16* wihB[3]; __bf16* whhB[3];
  for (int l = 0; l < 3; ++l) {
    wihB[l] = (__bf16*)alloc((size_t)3 * dout[l] * din[l]  * 2);
    whhB[l] = (__bf16*)alloc((size_t)3 * dout[l] * dout[l] * 2);
  }
  __bf16* x0 = (__bf16*)alloc((size_t)BSQ * EQ * 2);
  __bf16* y0 = (__bf16*)alloc((size_t)BSQ * HQ * 2);
  __bf16* y1 = (__bf16*)alloc((size_t)BSQ * HQ * 2);
  float*  xg = (float*) alloc((size_t)BSQ * 3 * HQ * 4);
  float*  hF = (float*) alloc((size_t)BQ * HQ * 4);
  __bf16* hB = (__bf16*)alloc((size_t)BQ * HQ * 2);

  // barrier state must be zeroed every call (ws is poisoned once to 0xAA)
  hipMemsetAsync(bar, 0, 256, stream);

  // weights -> bf16
  for (int l = 0; l < 3; ++l) {
    const long nih = (long)3 * dout[l] * din[l];
    const long nhh = (long)3 * dout[l] * dout[l];
    cvt_f32_bf16<<<2048, 256, 0, stream>>>(Wih[l], wihB[l], nih);
    cvt_f32_bf16<<<2048, 256, 0, stream>>>(Whh[l], whhB[l], nhh);
  }

  // embedding gather -> bf16
  embed_gather<<<4096, 256, 0, stream>>>(tokens, emb, x0, (long)BSQ * EQ);

  const __bf16* Xin[3] = {x0, y0, y1};
  __bf16*       Yb [3] = {y0, y1, nullptr};

  for (int l = 0; l < 3; ++l) {
    const int K = din[l], Ho = dout[l], N = 3 * Ho;

    // xg = Xin * W_ih^T + b_ih   (all 16384 rows at once)
    dim3 grid(BSQ / 64, N / 256);
    wmma_gemm_bias<<<grid, 256, 0, stream>>>(Xin[l], wihB[l], bih[l], xg, BSQ, N, K);

    // h0 = 0
    hipMemsetAsync(hF, 0, (size_t)BQ * Ho * 4, stream);
    hipMemsetAsync(hB, 0, (size_t)BQ * Ho * 2, stream);

    const int    nb   = Ho / 32;                                  // 64 / 64 / 32 blocks
    const size_t smem = (size_t)BQ * Ho * 2 + 3 * 32 * 32 * 4;    // h + hg tile
    (void)hipFuncSetAttribute((const void*)gru_scan,
                              hipFuncAttributeMaxDynamicSharedMemorySize,
                              (int)smem);
    gru_scan<<<nb, 256, smem, stream>>>(whhB[l], bhh[l], xg, hF, hB,
                                        Yb[l], (l == 2) ? out : nullptr,
                                        Ho, SQ, bar, (unsigned)nb);
  }
}
